// SelfSupervisedTrain_2259152797909
// MI455X (gfx1250) — matrix operand order repaired and checked by reading, not compiled, not run on hardware
//
#include <hip/hip_runtime.h>
#include <hip/hip_bf16.h>

// ---------------------------------------------------------------------------
// Types for CDNA5 WMMA (wave32): 16x16x32 bf16 -> f32 accumulate
// ---------------------------------------------------------------------------
typedef __attribute__((ext_vector_type(16))) __bf16 v16bf;
typedef __attribute__((ext_vector_type(8)))  float  v8f;

// fp32 -> bf16 (round to nearest even), bit-level
__device__ __forceinline__ unsigned short f2bf(float f) {
    unsigned int u = __float_as_uint(f);
    unsigned int r = (u + 0x7FFFu + ((u >> 16) & 1u)) >> 16;
    return (unsigned short)r;
}
__device__ __forceinline__ float bf2f(unsigned short b) {
    return __uint_as_float(((unsigned int)b) << 16);
}

// ---------------------------------------------------------------------------
// 1) zero a float buffer
// ---------------------------------------------------------------------------
__global__ void zero_kernel(float* __restrict__ p, int n) {
    int i = blockIdx.x * 256 + threadIdx.x;
    if (i < n) p[i] = 0.0f;
}

// ---------------------------------------------------------------------------
// 2) degree: deg[row[e]] += 1  (row = ei[0][e])
// ---------------------------------------------------------------------------
__global__ void deg_kernel(const int* __restrict__ ei, float* __restrict__ deg, int E) {
    int e = blockIdx.x * 256 + threadIdx.x;
    if (e < E) atomicAdd(&deg[ei[e]], 1.0f);
}

// ---------------------------------------------------------------------------
// 3) dinv = deg>0 ? rsqrt(max(deg,1)) : 0
// ---------------------------------------------------------------------------
__global__ void dinv_kernel(const float* __restrict__ deg, float* __restrict__ dinv, int n) {
    int i = blockIdx.x * 256 + threadIdx.x;
    if (i < n) {
        float d = deg[i];
        dinv[i] = (d > 0.0f) ? rsqrtf(fmaxf(d, 1.0f)) : 0.0f;
    }
}

// ---------------------------------------------------------------------------
// 4) edge scatter: tx1[col] += -(dinv[row]*dinv[col]) * x[row]   (C_IN=5)
// ---------------------------------------------------------------------------
__global__ void scatter_kernel(const int* __restrict__ ei, const float* __restrict__ x,
                               const float* __restrict__ dinv, float* __restrict__ tx1, int E) {
    int e = blockIdx.x * 256 + threadIdx.x;
    if (e >= E) return;
    int r = ei[e];
    int c = ei[E + e];
    float nrm = -(dinv[r] * dinv[c]);
    #pragma unroll
    for (int ch = 0; ch < 5; ++ch)
        atomicAdd(&tx1[c * 5 + ch], nrm * x[r * 5 + ch]);
}

// ---------------------------------------------------------------------------
// 5) node transform: H[n,c] = relu(x[n,:]@W0 + tx1[n,:]@W1 + b)[c]  -> bf16
//    H is [N,64] row-major == [1024, 3968] row-major after reshape.
// ---------------------------------------------------------------------------
__global__ __launch_bounds__(256)
void cheb_node_kernel(const float* __restrict__ x, const float* __restrict__ tx1,
                      const float* __restrict__ W0, const float* __restrict__ W1,
                      const float* __restrict__ b, unsigned short* __restrict__ H, int Nn) {
    __shared__ float sW0[5 * 64];
    __shared__ float sW1[5 * 64];
    __shared__ float sb[64];
    for (int i = threadIdx.x; i < 320; i += 256) { sW0[i] = W0[i]; sW1[i] = W1[i]; }
    if (threadIdx.x < 64) sb[threadIdx.x] = b[threadIdx.x];
    __syncthreads();
    int gid = blockIdx.x * 256 + threadIdx.x;
    if (gid >= Nn * 64) return;
    int node = gid >> 6, c = gid & 63;
    float acc = sb[c];
    #pragma unroll
    for (int k = 0; k < 5; ++k)
        acc += x[node * 5 + k] * sW0[k * 64 + c] + tx1[node * 5 + k] * sW1[k * 64 + c];
    acc = acc > 0.0f ? acc : 0.0f;
    H[gid] = f2bf(acc);
}

// ---------------------------------------------------------------------------
// 6) tiled transpose + fp32->bf16 convert:  W[K,N] f32  ->  WT[N,K] bf16
//    32x32 LDS tile, 256 threads (32x8), coalesced both sides.
// ---------------------------------------------------------------------------
__global__ __launch_bounds__(256)
void convert_transpose_kernel(const float* __restrict__ in, unsigned short* __restrict__ out,
                              int K, int N) {
    __shared__ unsigned short tile[32][33];
    int kb = blockIdx.x * 32, nb = blockIdx.y * 32;
    int tx = threadIdx.x & 31, ty = threadIdx.x >> 5;  // 32 x 8
    #pragma unroll
    for (int i = 0; i < 32; i += 8)
        tile[ty + i][tx] = f2bf(in[(size_t)(kb + ty + i) * N + nb + tx]);
    __syncthreads();
    #pragma unroll
    for (int i = 0; i < 32; i += 8)
        out[(size_t)(nb + ty + i) * K + kb + tx] = tile[tx][ty + i];
}

// ---------------------------------------------------------------------------
// 7) WMMA bf16 GEMM, LDS-free (fragments streamed from L2), fused BN+ReLU.
//    C[M,N] = relu( BN( A[M,K] @ BT[N,K]^T + bias ) ), bf16 out.
//    Block: 256 threads = 8 waves (4 M x 2 N), block tile 128x64.
//    Wave tile: 32x32 = 2 A-frags x 2 B-frags -> 4 WMMA per K-step of 32.
//    Per-lane fragment chunks are contiguous 16B pieces:
//      A row-major: lane m reads [kb+8h .. +7] and [kb+16+8h .. +7]
//      BT row-major: lane n reads [kb+8h .. +7] and [kb+16+8h .. +7]
// ---------------------------------------------------------------------------
__global__ __launch_bounds__(256)
void gemm_bn_relu_kernel(const unsigned short* __restrict__ A,
                         const unsigned short* __restrict__ BT,
                         const float* __restrict__ bias, const float* __restrict__ gam,
                         const float* __restrict__ bet,  const float* __restrict__ rm,
                         const float* __restrict__ rv,
                         unsigned short* __restrict__ C,
                         int M, int N, int K) {
    const int tid  = threadIdx.x;
    const int wave = tid >> 5;
    const int lane = tid & 31;
    const int half = lane >> 4;
    const int l16  = lane & 15;
    const int wm   = wave >> 1;   // 0..3
    const int wn   = wave & 1;    // 0..1
    const int rowBase = blockIdx.x * 128 + wm * 32;
    const int colBase = blockIdx.y * 64 + wn * 32;

    const unsigned short* a0 = A  + (size_t)(rowBase + l16) * K;
    const unsigned short* a1 = A  + (size_t)(rowBase + 16 + l16) * K;
    const unsigned short* b0 = BT + (size_t)(colBase + l16) * K;
    const unsigned short* b1 = BT + (size_t)(colBase + 16 + l16) * K;

    v8f acc00 = {}, acc01 = {}, acc10 = {}, acc11 = {};

    union Frag { v16bf v; uint4 q[2]; };

    #pragma unroll 2
    for (int kb = 0; kb < K; kb += 32) {
        Frag fa0, fa1, fb0, fb1;
        const int ka  = kb + 8 * half;    // A chunks: ka, ka+16
        const int kbb = kb + 8 * half;    // B chunks: kbb, kbb+16
        fa0.q[0] = *(const uint4*)(a0 + ka);
        fa0.q[1] = *(const uint4*)(a0 + ka + 16);
        fa1.q[0] = *(const uint4*)(a1 + ka);
        fa1.q[1] = *(const uint4*)(a1 + ka + 16);
        fb0.q[0] = *(const uint4*)(b0 + kbb);
        fb0.q[1] = *(const uint4*)(b0 + kbb + 16);
        fb1.q[0] = *(const uint4*)(b1 + kbb);
        fb1.q[1] = *(const uint4*)(b1 + kbb + 16);

        acc00 = __builtin_amdgcn_wmma_f32_16x16x32_bf16(false, fa0.v, false, fb0.v,
                                                        (short)0, acc00, false, false);
        acc01 = __builtin_amdgcn_wmma_f32_16x16x32_bf16(false, fa0.v, false, fb1.v,
                                                        (short)0, acc01, false, false);
        acc10 = __builtin_amdgcn_wmma_f32_16x16x32_bf16(false, fa1.v, false, fb0.v,
                                                        (short)0, acc10, false, false);
        acc11 = __builtin_amdgcn_wmma_f32_16x16x32_bf16(false, fa1.v, false, fb1.v,
                                                        (short)0, acc11, false, false);
    }

    // epilogue: C layout VGPR j -> M=j+8*half, lane -> N=l16 (+tile offsets)
    v8f accs[2][2] = { { acc00, acc01 }, { acc10, acc11 } };
    #pragma unroll
    for (int ni = 0; ni < 2; ++ni) {
        int col = colBase + ni * 16 + l16;
        float s  = gam[col] * rsqrtf(rv[col] + 1e-5f);
        float sh = (bias[col] - rm[col]) * s + bet[col];
        #pragma unroll
        for (int mi = 0; mi < 2; ++mi) {
            #pragma unroll
            for (int j = 0; j < 8; ++j) {
                int row = rowBase + mi * 16 + half * 8 + j;
                float val = accs[mi][ni][j] * s + sh;
                val = val > 0.0f ? val : 0.0f;
                C[(size_t)row * N + col] = f2bf(val);
            }
        }
    }
}

// ---------------------------------------------------------------------------
// 8) head: out[row,:] = finalize( H3[row,:] @ W3 + b3 )
//    mode 0 = softmax over dout; mode 1 = L2 normalize.
//    One block (128 threads) per row; H3 row staged in LDS.
// ---------------------------------------------------------------------------
__global__ __launch_bounds__(128)
void head_kernel(const unsigned short* __restrict__ H3, const float* __restrict__ W3,
                 const float* __restrict__ b3, float* __restrict__ out, int dout, int mode) {
    __shared__ float hrow[256];
    __shared__ float red[128];
    int row = blockIdx.x, t = threadIdx.x;
    for (int i = t; i < 256; i += 128) hrow[i] = bf2f(H3[row * 256 + i]);
    __syncthreads();

    float acc = 0.0f;
    if (t < dout) {
        acc = b3[t];
        for (int k = 0; k < 256; ++k) acc += hrow[k] * W3[k * dout + t];
    }

    if (mode == 0) {
        // softmax
        red[t] = (t < dout) ? acc : -3.0e38f;
        __syncthreads();
        for (int s = 64; s > 0; s >>= 1) {
            if (t < s) red[t] = fmaxf(red[t], red[t + s]);
            __syncthreads();
        }
        float mx = red[0];
        __syncthreads();
        float ex = (t < dout) ? __expf(acc - mx) : 0.0f;
        red[t] = ex;
        __syncthreads();
        for (int s = 64; s > 0; s >>= 1) {
            if (t < s) red[t] += red[t + s];
            __syncthreads();
        }
        float inv = 1.0f / red[0];
        if (t < dout) out[(size_t)row * dout + t] = ex * inv;
    } else {
        // L2 normalize
        red[t] = (t < dout) ? acc * acc : 0.0f;
        __syncthreads();
        for (int s = 64; s > 0; s >>= 1) {
            if (t < s) red[t] += red[t + s];
            __syncthreads();
        }
        float inv = 1.0f / fmaxf(sqrtf(red[0]), 1e-12f);
        if (t < dout) out[(size_t)row * dout + t] = acc * inv;
    }
}

// ---------------------------------------------------------------------------
// Host launcher
// ---------------------------------------------------------------------------
extern "C" void kernel_launch(void* const* d_in, const int* in_sizes, int n_in,
                              void* d_out, int out_size, void* d_ws, size_t ws_size,
                              hipStream_t stream) {
    const int N  = 63488;    // nodes
    const int E  = 4000000;  // edges
    const int Bn = 1024;     // batch (GEMM M)
    const int DF = 3968;     // D_FLAT (GEMM1 K)
    const int L1 = 512;      // GEMM1 N / GEMM2 K
    const int L2 = 256;      // GEMM2 N

    char* ws = (char*)d_ws;
    float*          deg  = (float*)(ws + 0);          //   253,952 B (N f32)
    float*          tx1  = (float*)(ws + 253952);     // 1,269,760 B (N*5 f32) (contig after deg)
    float*          dinv = (float*)(ws + 1523712);    //   253,952 B
    unsigned short* H    = (unsigned short*)(ws + 1777664);   // N*64 bf16 = [1024,3968]
    unsigned short* H2   = (unsigned short*)(ws + 9904128);   // 1024*512 bf16
    unsigned short* H3   = (unsigned short*)(ws + 10952704);  // 1024*256 bf16
    unsigned short* W1t  = (unsigned short*)(ws + 11476992);  // [512,3968] bf16 (W1^T)
    unsigned short* W2t  = (unsigned short*)(ws + 15540224);  // [256,512]  bf16 (W2^T)

    const float* cW0 = (const float*)d_in[8];
    const float* cW1 = (const float*)d_in[9];
    const float* cb  = (const float*)d_in[10];

    const int    headBase[3] = {11, 25, 39};           // HF, HS, PJ param blocks
    const int    brHead[4]   = {0, 1, 2, 2};
    const int    brDout[4]   = {5, 21, 128, 128};
    const int    brMode[4]   = {0, 0, 1, 1};
    const size_t brOff[4]    = {0, 5120, 26624, 157696};

    float* out = (float*)d_out;

    for (int br = 0; br < 4; ++br) {
        const float* x  = (const float*)d_in[2 * br];
        const int*   ei = (const int*)d_in[2 * br + 1];
        int hb = headBase[brHead[br]];
        const float* W1  = (const float*)d_in[hb + 0];
        const float* b1  = (const float*)d_in[hb + 1];
        const float* g1  = (const float*)d_in[hb + 2];
        const float* be1 = (const float*)d_in[hb + 3];
        const float* m1  = (const float*)d_in[hb + 4];
        const float* v1  = (const float*)d_in[hb + 5];
        const float* W2  = (const float*)d_in[hb + 6];
        const float* b2  = (const float*)d_in[hb + 7];
        const float* g2  = (const float*)d_in[hb + 8];
        const float* be2 = (const float*)d_in[hb + 9];
        const float* m2  = (const float*)d_in[hb + 10];
        const float* v2  = (const float*)d_in[hb + 11];
        const float* W3  = (const float*)d_in[hb + 12];
        const float* b3  = (const float*)d_in[hb + 13];

        // ---- ChebConv (memory/atomic bound; ~200MB traffic/branch) ----
        zero_kernel<<<(N * 6 + 255) / 256, 256, 0, stream>>>(deg, N * 6);  // deg + tx1
        deg_kernel<<<(E + 255) / 256, 256, 0, stream>>>(ei, deg, E);
        dinv_kernel<<<(N + 255) / 256, 256, 0, stream>>>(deg, dinv, N);
        scatter_kernel<<<(E + 255) / 256, 256, 0, stream>>>(ei, x, dinv, tx1, E);
        cheb_node_kernel<<<(N * 64 + 255) / 256, 256, 0, stream>>>(x, tx1, cW0, cW1, cb, H, N);

        // ---- stage head weights: fp32 [K,N] -> bf16 [N,K] (transposed) ----
        convert_transpose_kernel<<<dim3(DF / 32, L1 / 32), 256, 0, stream>>>(W1, W1t, DF, L1);
        convert_transpose_kernel<<<dim3(L1 / 32, L2 / 32), 256, 0, stream>>>(W2, W2t, L1, L2);

        // ---- MLP via LDS-free WMMA bf16 GEMMs (L2-resident working set) ----
        gemm_bn_relu_kernel<<<dim3(Bn / 128, L1 / 64), 256, 0, stream>>>(
            H, W1t, b1, g1, be1, m1, v1, H2, Bn, L1, DF);
        gemm_bn_relu_kernel<<<dim3(Bn / 128, L2 / 64), 256, 0, stream>>>(
            H2, W2t, b2, g2, be2, m2, v2, H3, Bn, L2, L1);

        // ---- final tiny GEMM + softmax / L2-norm ----
        head_kernel<<<Bn, 128, 0, stream>>>(H3, W3, b3, out + brOff[br], brDout[br], brMode[br]);
    }
}
